// BertMoEBlock_42691974922300
// MI455X (gfx1250) — matrix operand order repaired and checked by reading, not compiled, not run on hardware
//
#include <hip/hip_runtime.h>
#include <hip/hip_bf16.h>
#include <math.h>

#define NE    8
#define HID   1024
#define INTER 4096
#define NTOK  8192
#define CAP   NTOK
#define NT    32      // tokens per workgroup tile

typedef __attribute__((ext_vector_type(16))) __bf16 v16bf;
typedef __attribute__((ext_vector_type(8)))  __bf16 v8bf;
typedef __attribute__((ext_vector_type(4)))  __bf16 v4bf;
typedef __attribute__((ext_vector_type(8)))  float  v8f;

// ---------------- init: zero output + per-expert counters ----------------
__global__ void k_init(float* __restrict__ out, int n, int* __restrict__ cnt) {
  int i = blockIdx.x * blockDim.x + threadIdx.x;
  if (i < n) out[i] = 0.0f;
  if (i < NE) cnt[i] = 0;
}

// ---------------- one-shot fp32 -> bf16 weight conversion ----------------
__global__ void k_cvt(const float* __restrict__ src, __bf16* __restrict__ dst, long n4) {
  long stride = (long)gridDim.x * blockDim.x;
  for (long i = (long)blockIdx.x * blockDim.x + threadIdx.x; i < n4; i += stride) {
    float4 v = ((const float4*)src)[i];
    v4bf o;
    o[0] = (__bf16)v.x; o[1] = (__bf16)v.y; o[2] = (__bf16)v.z; o[3] = (__bf16)v.w;
    ((v4bf*)dst)[i] = o;
  }
}

// ---------------- router: logits -> softmax -> top2 -> expert lists ------
__global__ __launch_bounds__(256) void k_router(
    const float* __restrict__ X, const float* __restrict__ Wg,
    int* __restrict__ cnt, int* __restrict__ tokl, float* __restrict__ wl) {
  const int lane  = threadIdx.x & 31;
  const int wave  = threadIdx.x >> 5;
  const int token = blockIdx.x * 8 + wave;
  if (token >= NTOK) return;
  const float* x = X + (size_t)token * HID;

  float logit[NE];
  #pragma unroll
  for (int e = 0; e < NE; ++e) {
    const float* g = Wg + e * HID;
    float s = 0.f;
    for (int j = lane; j < HID; j += 32) s += x[j] * g[j];
    #pragma unroll
    for (int off = 16; off > 0; off >>= 1) s += __shfl_xor(s, off, 32);
    logit[e] = s;                       // full sum on every lane (xor tree)
  }
  float mx = logit[0];
  #pragma unroll
  for (int e = 1; e < NE; ++e) mx = fmaxf(mx, logit[e]);
  float p[NE], den = 0.f;
  #pragma unroll
  for (int e = 0; e < NE; ++e) { p[e] = expf(logit[e] - mx); den += p[e]; }
  const float inv = 1.f / den;
  #pragma unroll
  for (int e = 0; e < NE; ++e) p[e] *= inv;
  // top-2, first-index tie-break (matches lax.top_k)
  int i0 = 0;
  #pragma unroll
  for (int e = 1; e < NE; ++e) if (p[e] > p[i0]) i0 = e;
  int i1 = (i0 == 0) ? 1 : 0;
  #pragma unroll
  for (int e = 0; e < NE; ++e) if (e != i0 && p[e] > p[i1]) i1 = e;

  if (lane == 0) {
    int pos0 = atomicAdd(&cnt[i0], 1);
    tokl[i0 * CAP + pos0] = token;  wl[i0 * CAP + pos0] = p[i0];
    int pos1 = atomicAdd(&cnt[i1], 1);
    tokl[i1 * CAP + pos1] = token;  wl[i1 * CAP + pos1] = p[i1];
  }
}

// ---- A-fragment loaders: two contiguous 8-elem K chunks per lane --------
__device__ __forceinline__ v16bf loadA(const float* __restrict__ wr, int ka) {
  float4 f0 = *(const float4*)(wr + ka);
  float4 f1 = *(const float4*)(wr + ka + 4);
  float4 f2 = *(const float4*)(wr + ka + 16);
  float4 f3 = *(const float4*)(wr + ka + 20);
  v16bf a;
  a[0]  = (__bf16)f0.x; a[1]  = (__bf16)f0.y; a[2]  = (__bf16)f0.z; a[3]  = (__bf16)f0.w;
  a[4]  = (__bf16)f1.x; a[5]  = (__bf16)f1.y; a[6]  = (__bf16)f1.z; a[7]  = (__bf16)f1.w;
  a[8]  = (__bf16)f2.x; a[9]  = (__bf16)f2.y; a[10] = (__bf16)f2.z; a[11] = (__bf16)f2.w;
  a[12] = (__bf16)f3.x; a[13] = (__bf16)f3.y; a[14] = (__bf16)f3.z; a[15] = (__bf16)f3.w;
  return a;
}
__device__ __forceinline__ v16bf loadA(const __bf16* __restrict__ wr, int ka) {
  const v8bf lo = *(const v8bf*)(wr + ka);        // 16B global_load_b128
  const v8bf hh = *(const v8bf*)(wr + ka + 16);   // 16B global_load_b128
  return __builtin_shufflevector(lo, hh, 0,1,2,3,4,5,6,7,8,9,10,11,12,13,14,15);
}

// ---------------- fused expert FFN: up -> gelu -> down, 32-token tiles ----
template <typename WT>
__global__ __launch_bounds__(256) void k_ffn(
    const float* __restrict__ X,     const WT* __restrict__ Wup,
    const float* __restrict__ bup,   const WT* __restrict__ Wdown,
    const float* __restrict__ bdown, const int* __restrict__ cnt,
    const int* __restrict__ tokl,    const float* __restrict__ wl,
    float* __restrict__ out) {
  const int e    = blockIdx.y;
  const int tile = blockIdx.x;
  const int ce   = cnt[e];
  if (tile * NT >= ce) return;

  __shared__ __align__(32) __bf16 Xb[HID * NT];   // B-layout: [k][token]
  __shared__ __align__(32) __bf16 Hb[128 * NT];   // B-layout: [k_local][token]
  __shared__ __align__(16) float  Xs[NT * 512];   // async staging (half of X tile)
  __shared__ int   s_tok[NT];
  __shared__ float s_w[NT];

  const int t = threadIdx.x;
  if (t < NT) {
    int slot = tile * NT + t;
    if (slot < ce) { s_tok[t] = tokl[e * CAP + slot]; s_w[t] = wl[e * CAP + slot]; }
    else           { s_tok[t] = -1;                   s_w[t] = 0.f; }
  }
  __syncthreads();

  // stage gathered activations via async global->LDS DMA, in 2 halves,
  // then convert fp32 -> bf16 transposed into WMMA B-layout
  const unsigned ldsXs = (unsigned)(uintptr_t)(&Xs[0]);   // LDS byte offset
  for (int h = 0; h < 2; ++h) {
    for (int i = t; i < NT * 128; i += 256) {             // 16B per op
      int n  = i >> 7;                                    // token slot
      int c4 = (i & 127) * 4;                             // col within half
      int row = s_tok[n]; if (row < 0) row = 0;
      const float* gp = X + (size_t)row * HID + h * 512 + c4;
      unsigned lp = ldsXs + (unsigned)((n * 512 + c4) * 4);
      asm volatile("global_load_async_to_lds_b128 %0, %1, off"
                   :: "v"(lp), "v"(gp) : "memory");
    }
    asm volatile("s_wait_asynccnt 0" ::: "memory");
    __syncthreads();
    for (int i = t; i < NT * 128; i += 256) {
      int n  = i >> 7;
      int c4 = (i & 127) * 4;
      const float4 v = *(const float4*)&Xs[n * 512 + c4];
      const int c = h * 512 + c4;
      Xb[(c + 0) * NT + n] = (__bf16)v.x;
      Xb[(c + 1) * NT + n] = (__bf16)v.y;
      Xb[(c + 2) * NT + n] = (__bf16)v.z;
      Xb[(c + 3) * NT + n] = (__bf16)v.w;
    }
    __syncthreads();
  }

  const int lane  = t & 31;
  const int wave  = t >> 5;
  const int mlane = lane & 15;
  const int hi    = lane >> 4;

  v8f acc2[8][2];
  #pragma unroll
  for (int i = 0; i < 8; ++i)
    #pragma unroll
    for (int g = 0; g < 2; ++g)
      acc2[i][g] = (v8f){0.f,0.f,0.f,0.f,0.f,0.f,0.f,0.f};

  const WT* wupRow  = Wup   + ((size_t)e * INTER + wave * 16  + mlane) * HID;
  const WT* wdnBase = Wdown + ((size_t)e * HID   + wave * 128 + mlane) * INTER;

  for (int ic = 0; ic < INTER; ic += 128) {
    // ---- GEMM1: h^T tiles for inter rows [ic + wave*16, +16) ----
    v8f acc1[2];
    acc1[0] = (v8f){0.f,0.f,0.f,0.f,0.f,0.f,0.f,0.f};
    acc1[1] = (v8f){0.f,0.f,0.f,0.f,0.f,0.f,0.f,0.f};
    const WT* wr = wupRow + (size_t)ic * HID;
    #pragma unroll 4
    for (int k0 = 0; k0 < HID; k0 += 32) {
      const v16bf a = loadA(wr, k0 + hi * 8);     // one weight fragment ...
      #pragma unroll
      for (int g = 0; g < 2; ++g) {               // ... feeds 2 token groups
        v16bf b;
        #pragma unroll
        for (int s = 0; s < 16; ++s)
          b[s] = Xb[(k0 + hi * 8 + (s & 7) + (s >> 3) * 16) * NT + g * 16 + mlane];
        acc1[g] = __builtin_amdgcn_wmma_f32_16x16x32_bf16(
                      false, a, false, b, (short)0, acc1[g], false, false);
      }
    }
    // bias + exact GELU, scatter into Hb in B-layout
    #pragma unroll
    for (int v = 0; v < 8; ++v) {
      const int m  = v + 8 * hi;
      const float bb = bup[e * INTER + ic + wave * 16 + m];
      #pragma unroll
      for (int g = 0; g < 2; ++g) {
        float xg = acc1[g][v] + bb;
        float gl = 0.5f * xg * (1.0f + erff(xg * 0.70710678118654752f));
        Hb[(wave * 16 + m) * NT + g * 16 + mlane] = (__bf16)gl;
      }
    }
    __syncthreads();

    // ---- GEMM2: accumulate y^T over this 128-wide inter chunk ----
    #pragma unroll
    for (int t8 = 0; t8 < 8; ++t8) {
      const WT* wd = wdnBase + (size_t)(t8 * 16) * INTER + ic;
      #pragma unroll
      for (int kk = 0; kk < 4; ++kk) {
        const v16bf a = loadA(wd, kk * 32 + hi * 8);
        #pragma unroll
        for (int g = 0; g < 2; ++g) {
          v16bf b;
          #pragma unroll
          for (int s = 0; s < 16; ++s)
            b[s] = Hb[(kk * 32 + hi * 8 + (s & 7) + (s >> 3) * 16) * NT + g * 16 + mlane];
          acc2[t8][g] = __builtin_amdgcn_wmma_f32_16x16x32_bf16(
                            false, a, false, b, (short)0, acc2[t8][g], false, false);
        }
      }
    }
    __syncthreads();
  }

  // ---- epilogue: + bdown, scale by routing weight, combine across experts
  #pragma unroll
  for (int g = 0; g < 2; ++g) {
    const int   myTok = s_tok[g * 16 + mlane];
    const float myW   = s_w[g * 16 + mlane];
    if (myTok >= 0) {
      #pragma unroll
      for (int t8 = 0; t8 < 8; ++t8) {
        #pragma unroll
        for (int v = 0; v < 8; ++v) {
          const int hidx = wave * 128 + t8 * 16 + v + 8 * hi;
          float y = acc2[t8][g][v] + bdown[e * HID + hidx];
          atomicAdd(&out[(size_t)myTok * HID + hidx], myW * y);
        }
      }
    }
  }
}

extern "C" void kernel_launch(void* const* d_in, const int* in_sizes, int n_in,
                              void* d_out, int out_size, void* d_ws, size_t ws_size,
                              hipStream_t stream) {
  const float* X     = (const float*)d_in[0];
  const float* Wg    = (const float*)d_in[1];
  const float* Wup   = (const float*)d_in[2];
  const float* bup   = (const float*)d_in[3];
  const float* Wdown = (const float*)d_in[4];
  const float* bdown = (const float*)d_in[5];
  float* out = (float*)d_out;

  // workspace layout
  const size_t OFF_TOK = 4096;
  const size_t OFF_WL  = OFF_TOK + (size_t)NE * CAP * sizeof(int);
  const size_t OFF_WUP = 1u << 20;                                    // 1 MB
  const size_t WBYTES  = (size_t)NE * INTER * HID * sizeof(__bf16);   // 64 MB
  const size_t OFF_WDN = OFF_WUP + WBYTES;
  const size_t NEEDED  = OFF_WDN + WBYTES;

  int*    cnt  = (int*)d_ws;
  int*    tokl = (int*)((char*)d_ws + OFF_TOK);
  float*  wl   = (float*)((char*)d_ws + OFF_WL);
  __bf16* wupb = (__bf16*)((char*)d_ws + OFF_WUP);
  __bf16* wdnb = (__bf16*)((char*)d_ws + OFF_WDN);

  k_init<<<(out_size + 255) / 256, 256, 0, stream>>>(out, out_size, cnt);
  k_router<<<NTOK / 8, 256, 0, stream>>>(X, Wg, cnt, tokl, wl);

  dim3 grid(CAP / NT, NE);
  if (ws_size >= NEEDED) {
    const long n4 = (long)NE * INTER * HID / 4;
    k_cvt<<<4096, 256, 0, stream>>>(Wup,   wupb, n4);
    k_cvt<<<4096, 256, 0, stream>>>(Wdown, wdnb, n4);
    k_ffn<__bf16><<<grid, 256, 0, stream>>>(X, wupb, bup, wdnb, bdown,
                                            cnt, tokl, wl, out);
  } else {
    k_ffn<float><<<grid, 256, 0, stream>>>(X, Wup, bup, Wdown, bdown,
                                           cnt, tokl, wl, out);
  }
}
